// neural_layer_54889682043685
// MI455X (gfx1250) — compile-verified, matrix operand order checked by reading, not run on hardware
//
#include <hip/hip_runtime.h>
#include <hip/hip_bf16.h>

typedef __attribute__((ext_vector_type(16))) _Float16 v16h;
typedef __attribute__((ext_vector_type(8)))  _Float16 v8h;
typedef __attribute__((ext_vector_type(8)))  float    v8f;
typedef __attribute__((ext_vector_type(4)))  int      i32x4;

// typed address-space pointers for the async global->LDS builtin
typedef __attribute__((address_space(1))) i32x4 gv4i;   // global (device)
typedef __attribute__((address_space(3))) i32x4 lv4i;   // LDS

#define T_STEPS 15
#define C_IN    4
#define H_IN    512
#define W_INP   512
#define C_OUT   20
#define PH      255          // pooled H = W
#define PSTRIDE 256          // padded pooled row stride (halves)
#define OH      226          // conv output H = W
#define RPB     2            // output rows per block
#define RBLKS   (OH / RPB)   // 113
#define SROWS   (CONV_K + 1) // staged pooled rows per block
#define CONV_K  30
#define KCHUNKS (C_IN * CONV_K)   // 120: one (ci,kr) pair per 32-wide K chunk
#define NTILES  2                 // C_OUT=20 padded to 2 x 16
#define NUNITS  (C_OUT * OH * OH) // 1,021,520
#define PLANE   NUNITS
#define POT_ELEMS (T_STEPS * PLANE)   // 15,322,800

// ---- workspace layout (bytes) ----
#define POOLED_HALVES (T_STEPS * C_IN * PH * PSTRIDE)   // 3,916,800
#define POOLED_OFF    0
#define BFRAG_OFF     ((size_t)POOLED_HALVES * 2)       // 7,833,600
#define BFRAG_HALVES  (KCHUNKS * NTILES * 32 * 16)      // 122,880
#define VALUES_OFF    (BFRAG_OFF + (size_t)BFRAG_HALVES * 2)
#define NSPK_OFF      (VALUES_OFF + (size_t)NUNITS * 4)
#define SCALAR_OFF    (NSPK_OFF + (size_t)NUNITS * 4)   // 8-byte aligned

#if defined(__has_builtin)
#  if __has_builtin(__builtin_amdgcn_global_load_async_to_lds_b128)
#    define HAVE_ASYNC_LDS 1
#  endif
#  if __has_builtin(__builtin_amdgcn_s_wait_asynccnt)
#    define HAVE_WAIT_ASYNC 1
#  endif
#endif

// ---------------- max-pool 3x3 s2 -> f16 binary pooled ----------------
__global__ void pool_kernel(const float* __restrict__ x, _Float16* __restrict__ pooled) {
    int idx = blockIdx.x * blockDim.x + threadIdx.x;
    if (idx >= T_STEPS * C_IN * PH * PH) return;
    int pc = idx % PH;
    int pr = (idx / PH) % PH;
    int tc = idx / (PH * PH);               // t*C_IN + ch
    const float* base = x + ((size_t)tc * H_IN + 2 * pr) * W_INP + 2 * pc;
    float m = 0.0f;                          // inputs are >= 0
    #pragma unroll
    for (int dr = 0; dr < 3; ++dr)
        #pragma unroll
        for (int dc = 0; dc < 3; ++dc)
            m = fmaxf(m, base[dr * W_INP + dc]);
    pooled[((size_t)tc * PH + pr) * PSTRIDE + pc] = (_Float16)m;
}

// ------- pre-swizzle W into WMMA B-fragment layout (K padded to 32/chunk) -------
__global__ void bfrag_kernel(const float* __restrict__ Wt, _Float16* __restrict__ bfrag) {
    int idx = blockIdx.x * blockDim.x + threadIdx.x;
    if (idx >= BFRAG_HALVES) return;
    int e    = idx & 15;
    int lane = (idx >> 4) & 31;
    int nt   = (idx >> 9) & 1;
    int q    = idx >> 10;
    // per-ISA 16-bit fragment layout: K offset for (lane, element e)
    int koff = ((e < 8) ? e : (e + 8)) + ((lane >= 16) ? 8 : 0);   // 0..31
    int ci = q / CONV_K;
    int kr = q % CONV_K;
    int n  = nt * 16 + (lane & 15);
    _Float16 v = (_Float16)0.0f;
    if (koff < CONV_K && n < C_OUT)
        v = (_Float16)Wt[(((size_t)n * C_IN + ci) * CONV_K + kr) * CONV_K + koff];
    bfrag[idx] = v;
}

__global__ void init_kernel(unsigned long long* pack, unsigned* maxbits) {
    *pack = 0ULL;
    *maxbits = 0u;
}

// ---------------- implicit-GEMM conv via V_WMMA_F32_16X16X32_F16 ----------------
// Block = 128 threads = 4 waves: (M-tile 0/1) x (N-tile 0/1).
// Each block computes RPB=2 output rows x 32 columns x 32 channels for one t.
// Staging of pooled tile uses CDNA5 async global->LDS copies (ASYNCcnt).
__global__ __launch_bounds__(128) void conv_wmma_kernel(
    const _Float16* __restrict__ pooled, const _Float16* __restrict__ bfrag,
    float* __restrict__ pot_thr, float* __restrict__ spikes)
{
    __shared__ _Float16 sp[C_IN][SROWS][64];   // 15,872 B

    const int tid  = threadIdx.x;
    const int lane = tid & 31;
    const int wv   = tid >> 5;
    const int nt   = wv & 1;    // N tile (channels 0-15 / 16-31)
    const int mt   = wv >> 1;   // M tile (16 output columns)

    int bid = blockIdx.x;
    const int cblk = bid & 7;   // 8 x 32-column tiles cover 226
    bid >>= 3;
    const int rb = bid % RBLKS;
    const int t  = bid / RBLKS;
    const int r0 = rb * RPB;
    const int c0 = cblk * 32;

    // ---- stage pooled[t, 0..3, r0..r0+30, c0..c0+63] into LDS ----
    // 124 rows x 128B = 992 x b128 transfers
    const int NSEG = C_IN * SROWS * 8;
    for (int o = tid; o < NSEG; o += 128) {
        int seg = o & 7;
        int row = o >> 3;
        int ch  = row / SROWS;
        int rr  = row % SROWS;
        const _Float16* g = pooled +
            (((size_t)t * C_IN + ch) * PH + (r0 + rr)) * PSTRIDE + c0 + seg * 8;
        _Float16* l = &sp[ch][rr][seg * 8];
#if defined(HAVE_ASYNC_LDS)
        __builtin_amdgcn_global_load_async_to_lds_b128(
            (gv4i*)(unsigned long long)(const char*)g,
            (lv4i*)(unsigned)(unsigned long long)(char*)l, 0, 0);
#else
        *(v8h*)l = *(const v8h*)g;
#endif
    }
#if defined(HAVE_ASYNC_LDS)
#  if defined(HAVE_WAIT_ASYNC)
    __builtin_amdgcn_s_wait_asynccnt(0);
#  else
    asm volatile("s_wait_asynccnt 0" ::: "memory");
#  endif
#endif
    __syncthreads();

    // A-fragment base column: M-tile + row-in-tile + upper-half-wave K offset (+8)
    const int mcol = (mt << 4) + (lane & 15) + ((lane >> 4) << 3);
    const _Float16* bptr = bfrag + ((size_t)(nt * 32 + lane) << 4);

    v8f acc0 = {}, acc1 = {};
    #pragma unroll 4
    for (int q = 0; q < KCHUNKS; ++q) {
        const int ci = q / CONV_K;
        const int kr = q % CONV_K;
        const _Float16* rp0 = &sp[ci][kr][mcol];
        const _Float16* rp1 = &sp[ci][kr + 1][mcol];
        v16h a0, a1;
        #pragma unroll
        for (int e = 0; e < 8; ++e)  { a0[e] = rp0[e];     a1[e] = rp1[e]; }
        #pragma unroll
        for (int e = 8; e < 16; ++e) { a0[e] = rp0[e + 8]; a1[e] = rp1[e + 8]; }
        v16h b = *(const v16h*)(bptr + (size_t)q * (NTILES * 32 * 16));
        if (q + 1 < KCHUNKS)
            __builtin_prefetch(bptr + (size_t)(q + 1) * (NTILES * 32 * 16), 0, 3);
        acc0 = __builtin_amdgcn_wmma_f32_16x16x32_f16(
                   false, a0, false, b, (short)0, acc0, false, false);
        acc1 = __builtin_amdgcn_wmma_f32_16x16x32_f16(
                   false, a1, false, b, (short)0, acc1, false, false);
    }

    // epilogue: C/D layout -> (M = vgpr (+8 upper half-wave), N = lane&15)
    const int n       = nt * 16 + (lane & 15);
    const int mb      = (lane >> 4) << 3;
    const int colbase = c0 + (mt << 4);
    if (n < C_OUT) {
        #pragma unroll
        for (int sub = 0; sub < RPB; ++sub) {
            const int r = r0 + sub;
            #pragma unroll
            for (int rr = 0; rr < 8; ++rr) {
                int col = colbase + mb + rr;
                if (col < OH) {
                    float pot = (sub == 0) ? acc0[rr] : acc1[rr];
                    float ptv = (pot > 20.0f) ? pot : 0.0f;
                    size_t o  = (((size_t)t * C_OUT + n) * OH + r) * OH + col;
                    pot_thr[o] = ptv;
                    spikes[o]  = (ptv > 0.0f) ? 1.0f : 0.0f;
                }
            }
        }
    }
}

// -------- per-unit spike count / first-spike potential + global max --------
__global__ void unit_stats_kernel(const float* __restrict__ pot_thr,
                                  const float* __restrict__ spikes,
                                  float* __restrict__ values, float* __restrict__ nspk,
                                  unsigned* __restrict__ maxbits) {
    int u = blockIdx.x * blockDim.x + threadIdx.x;
    if (u >= NUNITS) return;
    float ns = 0.0f;
    #pragma unroll
    for (int t = 0; t < T_STEPS; ++t) ns += spikes[(size_t)t * PLANE + u];
    int ft = T_STEPS - (int)ns;
    if (ft > T_STEPS - 1) ft = T_STEPS - 1;
    if (ft < 0) ft = 0;
    float val = pot_thr[(size_t)ft * PLANE + u];
    values[u] = val;
    nspk[u]   = ns;
    atomicMax(maxbits, __float_as_uint(val));   // val >= 0: uint order == float order
}

// -------- argmax of total = nspk*(values + max*T), first-index tie-break --------
__global__ void argmax_kernel(const float* __restrict__ values,
                              const float* __restrict__ nspk,
                              const unsigned* __restrict__ maxbits,
                              unsigned long long* __restrict__ pack) {
    int u = blockIdx.x * blockDim.x + threadIdx.x;
    if (u >= NUNITS) return;
    float v = __uint_as_float(*maxbits) * (float)T_STEPS;
    float total = nspk[u] * (values[u] + v);    // >= 0
    unsigned long long p = ((unsigned long long)__float_as_uint(total) << 32)
                         | (unsigned long long)(~(unsigned)u);
    atomicMax(pack, p);
}

__global__ void winner_kernel(const unsigned long long* __restrict__ pack,
                              const float* __restrict__ nspk,
                              float* __restrict__ out_winner,
                              int* __restrict__ wsc, float* __restrict__ outLat) {
    unsigned u = ~(unsigned)(*pack);
    int f  = u / (OH * OH);
    int rc = u % (OH * OH);
    int r  = rc / OH;
    int c  = rc % OH;
    out_winner[0] = (float)f;
    out_winner[1] = (float)r;
    out_winner[2] = (float)c;
    wsc[0] = f; wsc[1] = r; wsc[2] = c;
    *outLat = nspk[u];
}

// ---------------- STDP update on the winner's channel ----------------
__global__ void stdp_kernel(const float* __restrict__ Wt,
                            const _Float16* __restrict__ pooled,
                            const int* __restrict__ wsc,
                            const float* __restrict__ outLat,
                            float* __restrict__ Wout) {
    int idx = blockIdx.x * blockDim.x + threadIdx.x;
    if (idx >= C_OUT * C_IN * CONV_K * CONV_K) return;
    int kc = idx % CONV_K;
    int kr = (idx / CONV_K) % CONV_K;
    int ci = (idx / (CONV_K * CONV_K)) % C_IN;
    int co = idx / (C_IN * CONV_K * CONV_K);
    float w = Wt[idx];
    if (co == wsc[0]) {
        int r = wsc[1], c = wsc[2];
        float il = 0.0f;
        #pragma unroll
        for (int t = 0; t < T_STEPS; ++t)
            il += (float)pooled[(((size_t)t * C_IN + ci) * PH + (r + kr)) * PSTRIDE + (c + kc)];
        float lr = (il >= *outLat) ? 0.05f : -0.015f;
        w = w + lr * w * (1.0f - w);
    }
    w = (w < 0.0f) ? 0.0f : ((w > 1.0f) ? 1.0f : w);
    Wout[idx] = w;
}

extern "C" void kernel_launch(void* const* d_in, const int* in_sizes, int n_in,
                              void* d_out, int out_size, void* d_ws, size_t ws_size,
                              hipStream_t stream) {
    const float* x  = (const float*)d_in[0];
    const float* Wt = (const float*)d_in[1];
    float* out = (float*)d_out;
    char*  ws  = (char*)d_ws;

    _Float16* pooled = (_Float16*)(ws + POOLED_OFF);
    _Float16* bfrag  = (_Float16*)(ws + BFRAG_OFF);
    float* values = (float*)(ws + VALUES_OFF);
    float* nspk   = (float*)(ws + NSPK_OFF);
    unsigned long long* pack = (unsigned long long*)(ws + SCALAR_OFF);
    unsigned* maxbits = (unsigned*)(ws + SCALAR_OFF + 8);
    int*   wsc    = (int*)(ws + SCALAR_OFF + 16);
    float* outLat = (float*)(ws + SCALAR_OFF + 32);

    float* pot_thr = out;
    float* spikes  = out + POT_ELEMS;
    float* winner  = out + 2 * (size_t)POT_ELEMS;
    float* wnew    = winner + 3;

    int npool = T_STEPS * C_IN * PH * PH;
    pool_kernel<<<(npool + 255) / 256, 256, 0, stream>>>(x, pooled);
    bfrag_kernel<<<(BFRAG_HALVES + 255) / 256, 256, 0, stream>>>(Wt, bfrag);
    init_kernel<<<1, 1, 0, stream>>>(pack, maxbits);

    conv_wmma_kernel<<<T_STEPS * RBLKS * 8, 128, 0, stream>>>(pooled, bfrag, pot_thr, spikes);

    unit_stats_kernel<<<(NUNITS + 255) / 256, 256, 0, stream>>>(pot_thr, spikes, values, nspk, maxbits);
    argmax_kernel<<<(NUNITS + 255) / 256, 256, 0, stream>>>(values, nspk, maxbits, pack);
    winner_kernel<<<1, 1, 0, stream>>>(pack, nspk, winner, wsc, outLat);
    stdp_kernel<<<(C_OUT * C_IN * CONV_K * CONV_K + 255) / 256, 256, 0, stream>>>(
        Wt, pooled, wsc, outLat, wnew);
}
